// Model_69363721830718
// MI455X (gfx1250) — compile-verified, matrix operand order checked by reading, not run on hardware
//
#include <hip/hip_runtime.h>
#include <math.h>

// ---------------------------------------------------------------------------
// MI455X (gfx1250) implementation.
//  - All GEMM-shaped work -> v_wmma_f32_16x16x32_bf16 (wave32, 16x16x32 tiles)
//  - bf16 conversion happens during LDS staging (no extra HBM passes)
//  - A-tile + impulse staging use global_load_async_to_lds_b128 (ASYNCcnt)
//  - FIR convolutions tiled through LDS, fp32 VALU
//  - generate_mix proven to be identity (softmax over singleton axis) -> skipped
// ---------------------------------------------------------------------------

typedef __attribute__((ext_vector_type(16))) __bf16 v16bf;
typedef __attribute__((ext_vector_type(8)))  float  v8f;

__device__ __forceinline__ unsigned short f2bfu(float f) {
  union { float f; unsigned int u; } c; c.f = f;
  unsigned int u = c.u;
  return (unsigned short)((u + 0x7FFFu + ((u >> 16) & 1u)) >> 16);
}

__device__ __forceinline__ v16bf frag_from(const unsigned short* p0, const unsigned short* p1) {
  union { uint4 q[2]; v16bf v; } u;
  u.q[0] = *(const uint4*)p0;
  u.q[1] = *(const uint4*)p1;
  return u.v;
}

// async global->LDS 16B copy (gfx1250, tracked by ASYNCcnt)
__device__ __forceinline__ void async_b128(void* lds_dst, const void* gsrc) {
  unsigned lds = (unsigned)(unsigned long long)lds_dst;          // LDS aperture: addr[31:0]
  unsigned long long g = (unsigned long long)gsrc;
  asm volatile("global_load_async_to_lds_b128 %0, %1, off" :: "v"(lds), "v"(g) : "memory");
}
__device__ __forceinline__ void async_wait0() {
  asm volatile("s_wait_asynccnt 0" ::: "memory");
}

// ---------------------------------------------------------------------------
// Generic bf16 WMMA GEMM: C(MxN,f32) = A(MxK,bf16 rowmajor) * B + optional bias
// BMODE: 0 = B f32 rowmajor [k*bp0 + n]
//        1 = conv weight (cout,cin,ktap): B[(n)*bp0*bp1 + (k%bp0)*bp1 + k/bp0]
//        2 = transposed weight (out,in):  B[n*bp0 + k]
//        3 = synthesized DFT matrix, bp0 = FFT size, fsc = 2*pi/bp0
// epi: 0 none, 1 bias, 2 bias+leaky(0.2)
// M,N multiples of 128; K multiple of 32.
// ---------------------------------------------------------------------------
template<int BMODE>
__global__ __launch_bounds__(256) void gemm_bf16(
    const unsigned short* __restrict__ A, const float* __restrict__ B,
    float* __restrict__ C, const float* __restrict__ bias,
    int M, int N, int K, int bp0, int bp1, int epi, float fsc)
{
  __shared__ unsigned short lA[128 * 40];
  __shared__ unsigned short lB[128 * 40];
  const int n0 = blockIdx.x * 128;
  const int m0 = blockIdx.y * 128;
  const int tid = threadIdx.x;
  const int wave = tid >> 5, lane = tid & 31;
  const int wm = wave >> 2, wn = wave & 3;       // 2x4 wave grid, 64x32 per wave
  const int half = lane >> 4, lm = lane & 15;
  const int koff = half * 8;

  v8f acc[4][2];
  v8f vz = {0.f,0.f,0.f,0.f,0.f,0.f,0.f,0.f};
  for (int i = 0; i < 4; ++i) for (int j = 0; j < 2; ++j) acc[i][j] = vz;

  const int ar = tid >> 1;
  const int ah = (tid & 1) << 4;

  for (int k0 = 0; k0 < K; k0 += 32) {
    { // stage A tile 128x32 bf16 via async global->LDS (2x16B per thread)
      const unsigned short* src = A + (size_t)(m0 + ar) * K + (k0 + ah);
      async_b128(&lA[ar * 40 + ah],     src);
      async_b128(&lA[ar * 40 + ah + 8], src + 8);
    }
    if (BMODE == 0) { // coalesced along n, transpose into lB[n][k]
      int kk = tid >> 3;
      int nb = (tid & 7) << 4;
      const float* src = B + (size_t)(k0 + kk) * bp0 + (n0 + nb);
      #pragma unroll
      for (int i = 0; i < 16; ++i) lB[(nb + i) * 40 + kk] = f2bfu(src[i]);
    } else {
      int n  = tid >> 1;
      int kh = (tid & 1) << 4;
      #pragma unroll
      for (int i = 0; i < 16; ++i) {
        int k  = k0 + kh + i;
        int nn = n0 + n;
        float v;
        if (BMODE == 1)      { int cin = k % bp0; int tap = k / bp0;
                               v = B[(size_t)nn * bp0 * bp1 + (size_t)cin * bp1 + tap]; }
        else if (BMODE == 2) { v = B[(size_t)nn * bp0 + k]; }
        else { // DFT: cols [0,bp0/2) cos, [bp0/2,bp0) sin (sign irrelevant for |.|)
          int bin = nn & ((bp0 >> 1) - 1);
          int t   = (k * bin) & (bp0 - 1);
          float ang = (float)t * fsc;
          v = (nn < (bp0 >> 1)) ? __cosf(ang) : __sinf(ang);
        }
        lB[n * 40 + kh + i] = f2bfu(v);
      }
    }
    async_wait0();
    __syncthreads();

    v16bf af[4], bfr[2];
    #pragma unroll
    for (int i = 0; i < 4; ++i) {
      int row = wm * 64 + i * 16 + lm;
      af[i] = frag_from(&lA[row * 40 + koff], &lA[row * 40 + 16 + koff]);
    }
    #pragma unroll
    for (int j = 0; j < 2; ++j) {
      int row = wn * 32 + j * 16 + lm;
      bfr[j] = frag_from(&lB[row * 40 + koff], &lB[row * 40 + 16 + koff]);
    }
    #pragma unroll
    for (int i = 0; i < 4; ++i)
      #pragma unroll
      for (int j = 0; j < 2; ++j)
        acc[i][j] = __builtin_amdgcn_wmma_f32_16x16x32_bf16(
            false, af[i], false, bfr[j], (short)0, acc[i][j], false, false);
    __syncthreads();
  }

  #pragma unroll
  for (int i = 0; i < 4; ++i)
    #pragma unroll
    for (int j = 0; j < 2; ++j) {
      int n = n0 + wn * 32 + j * 16 + lm;
      float bv = (epi >= 1 && bias) ? bias[n] : 0.0f;
      #pragma unroll
      for (int r = 0; r < 8; ++r) {
        int m = m0 + wm * 64 + i * 16 + half * 8 + r;
        float v = acc[i][j][r] + bv;
        if (epi == 2 && v < 0.f) v *= 0.2f;
        C[(size_t)m * N + n] = v;
      }
    }
}

// ------------------------- small / elementwise kernels ----------------------

__global__ void k_build_frames(const float* __restrict__ x, unsigned short* __restrict__ A) {
  int i = blockIdx.x * 256 + threadIdx.x;
  if (i >= 1024 * 2048) return;
  int r = i >> 11, c = i & 2047;
  int b = r >> 7,  f = r & 127;
  int idx = f * 256 + c;
  float v = (idx < 32768) ? x[(size_t)b * 32768 + idx] : 0.f;
  float w = 0.5f - 0.5f * __cosf(6.28318530717958647f * (float)c * (1.0f / 2048.0f));
  A[i] = f2bfu(v * w);
}

__global__ void k_mag(const float* __restrict__ spec2, float* __restrict__ act) {
  int i = blockIdx.x * 256 + threadIdx.x;
  if (i >= 1024 * 1024) return;
  int r = i >> 10, c = i & 1023;
  float re = spec2[(size_t)r * 2048 + c];
  float im = spec2[(size_t)r * 2048 + 1024 + c];
  act[i] = sqrtf(re * re + im * im);
}

__global__ void k_im2col_enc(const float* __restrict__ act, unsigned short* __restrict__ A, int d) {
  int i = blockIdx.x * 256 + threadIdx.x;
  if (i >= 1024 * 3072) return;
  int r = i / 3072, rem = i % 3072;
  int tap = rem >> 10, c = rem & 1023;
  int b = r >> 7, t = r & 127;
  int tt = t + tap * d;
  float v = (tt < 128) ? act[((size_t)b * 128 + tt) * 1024 + c] : 0.f;
  A[i] = f2bfu(v);
}

__global__ void k_bn_stats(const float* __restrict__ X, float* __restrict__ bn, int M, int N) {
  __shared__ float s1[256], s2[256];
  int c = blockIdx.x, tid = threadIdx.x;
  float a = 0.f, b = 0.f;
  for (int r = tid; r < M; r += 256) { float v = X[(size_t)r * N + c]; a += v; b += v * v; }
  s1[tid] = a; s2[tid] = b; __syncthreads();
  for (int s = 128; s > 0; s >>= 1) {
    if (tid < s) { s1[tid] += s1[tid + s]; s2[tid] += s2[tid + s]; }
    __syncthreads();
  }
  if (tid == 0) { float m = s1[0] / M; bn[c] = m; bn[N + c] = s2[0] / M - m * m; }
}

__global__ void k_bn_norm(float* __restrict__ X, const float* __restrict__ bn,
                          const float* __restrict__ g, const float* __restrict__ b,
                          int M, int N, int post_leaky) {
  int i = blockIdx.x * 256 + threadIdx.x;
  if (i >= M * N) return;
  int c = i % N;
  float m = bn[c], v = bn[N + c];
  float y = (X[i] - m) * rsqrtf(v + 1e-5f) * g[c] + b[c];
  if (post_leaky && y < 0.f) y *= 0.2f;
  X[i] = y;
}

__global__ void k_f2bf(const float* __restrict__ x, unsigned short* __restrict__ y, int n) {
  int i = blockIdx.x * 256 + threadIdx.x;
  if (i < n) y[i] = f2bfu(x[i]);
}

__global__ void k_fir(const float* __restrict__ in, float* __restrict__ out) {
  // ref[j] = (1 - j/7)^10, compile-time constants
  const float REF[8] = {1.0f, 0.2140628f, 0.0345714f, 0.0037125f,
                        2.09042e-4f, 3.62512e-6f, 3.54013e-9f, 0.0f};
  int i = blockIdx.x * 256 + threadIdx.x;
  if (i >= 1024 * 4096) return;
  int r = i >> 12, c = i & 4095;
  int b = r >> 7, t = r & 127;
  float acc = 0.f;
  #pragma unroll
  for (int j = 0; j < 7; ++j) {
    if (j <= t) acc += in[((size_t)b * 128 + (t - j)) * 4096 + c] * REF[j];
  }
  out[i] = acc;
}

__global__ void k_ctxt_mean(const float* __restrict__ dense, float* __restrict__ ctxt) {
  int i = blockIdx.x * 256 + threadIdx.x;
  if (i >= 8 * 4096) return;
  int b = i >> 12, c = i & 4095;
  float a = 0.f;
  for (int t = 0; t < 128; ++t) a += dense[((size_t)b * 128 + t) * 4096 + c];
  ctxt[i] = a * (1.0f / 128.0f);
}

__global__ void k_gemv(const float* __restrict__ W, const float* __restrict__ in,
                       const float* __restrict__ bias, float* __restrict__ out,
                       int Bn, int I, int O) {
  int i = blockIdx.x * 256 + threadIdx.x;
  if (i >= Bn * O) return;
  int b = i / O, o = i % O;
  float a = bias ? bias[o] : 0.f;
  const float* w = W + (size_t)o * I;
  const float* x = in + (size_t)b * I;
  for (int k = 0; k < I; ++k) a += w[k] * x[k];
  out[i] = a;
}

__global__ __launch_bounds__(1024) void k_topk(float* __restrict__ dense,
                                               int* __restrict__ ev_idx, float* __restrict__ ev_val) {
  __shared__ float sv[1024]; __shared__ int si[1024];
  int b = blockIdx.x, tid = threadIdx.x;
  for (int e = 0; e < 32; ++e) {
    float best = -3e38f; int bi = 0;
    for (int f = tid; f < 524288; f += 1024) {
      int t = f & 127, c = f >> 7;
      float v = dense[((size_t)b * 128 + t) * 4096 + c];
      if (v > best) { best = v; bi = f; }
    }
    sv[tid] = best; si[tid] = bi; __syncthreads();
    for (int s = 512; s > 0; s >>= 1) {
      if (tid < s && sv[tid + s] > sv[tid]) { sv[tid] = sv[tid + s]; si[tid] = si[tid + s]; }
      __syncthreads();
    }
    if (tid == 0) {
      int f = si[0];
      ev_idx[b * 32 + e] = f; ev_val[b * 32 + e] = sv[0];
      dense[((size_t)b * 128 + (f & 127)) * 4096 + (f >> 7)] = -3e38f;
    }
    __syncthreads();
  }
}

__global__ void k_zero(float* p, int n) {
  int i = blockIdx.x * 256 + threadIdx.x;
  if (i < n) p[i] = 0.f;
}

__global__ void k_scatter_enc(const int* __restrict__ ev_idx, const float* __restrict__ ev_val,
                              float* __restrict__ enc) {
  int i = threadIdx.x;
  if (i >= 256) return;
  int b = i >> 5;
  enc[(size_t)b * 524288 + ev_idx[i]] = ev_val[i];
}

__global__ void k_emb(const float* __restrict__ ce, const float* __restrict__ eo_w,
                      const float* __restrict__ eo_b, const int* __restrict__ ev_idx,
                      float* __restrict__ emb) {
  int i = blockIdx.x * 256 + threadIdx.x;
  if (i >= 256 * 256) return;
  int n = i >> 8, j = i & 255;
  int b = n >> 5;
  int ch = ev_idx[n] >> 7;
  emb[i] = ce[b * 256 + j] + eo_w[(size_t)j * 4096 + ch] + eo_b[j];
}

// im2col for tiny transposed-conv layers. layout0: src[n*512 + c*4 + ts] (from linear),
// layout1: src[(n*Tin + ts)*128 + c]. rep=1 -> input is 2x time-repeat of src.
__global__ void k_tf_im2col(const float* __restrict__ src, unsigned short* __restrict__ A,
                            int Tin, int Tout, int rep, int layout) {
  int M = 256 * Tout;
  int i = blockIdx.x * 256 + threadIdx.x;
  if (i >= M * 384) return;
  int r = i / 384, rem = i % 384;
  int tap = rem >> 7, c = rem & 127;
  int n = r / Tout, t = r % Tout;
  int tp = t - 1 + tap;
  float v = 0.f;
  if (tp >= 0 && tp < Tout) {
    int ts = rep ? (tp >> 1) : tp;
    v = layout ? src[((size_t)n * Tin + ts) * 128 + c] : src[(size_t)n * 512 + c * 4 + ts];
  }
  A[i] = f2bfu(v);
}

// nm_conv collapsed to 3 groups (interior/left-edge/right-edge of each 32x repeat block)
__global__ void k_nm_im2col(const float* __restrict__ src, unsigned short* __restrict__ A) {
  int i = blockIdx.x * 256 + threadIdx.x;
  if (i >= 12288 * 384) return;
  int r = i / 384, rem = i % 384;
  int tap = rem >> 7, c = rem & 127;
  int g = r >> 12, rr = r & 4095;
  int n = rr >> 4, k = rr & 15;
  int ks = k; bool pad = false;
  if (g == 1)      { if (tap == 0) { if (k == 0)  pad = true; else ks = k - 1; } }
  else if (g == 2) { if (tap == 2) { if (k == 15) pad = true; else ks = k + 1; } }
  float v = pad ? 0.f : src[((size_t)n * 16 + ks) * 128 + c];
  A[i] = f2bfu(v);
}

__global__ void k_nmspec(const float* __restrict__ h, const float* __restrict__ w,
                         const float* __restrict__ b, float* __restrict__ filt) {
  int r = blockIdx.x * 256 + threadIdx.x;
  if (r >= 12288) return;
  const float* hr = h + (size_t)r * 128;
  for (int bin = 0; bin < 5; ++bin) {
    float a = b[bin];
    const float* wb = w + bin * 128;
    for (int c = 0; c < 128; ++c) a += wb[c] * hr[c];
    filt[r * 5 + bin] = a * a;
  }
}

__global__ void k_nspec(const float* __restrict__ noise, float* __restrict__ nspec) {
  int f = blockIdx.x * 256 + threadIdx.x;
  if (f >= 512) return;
  for (int bin = 0; bin < 5; ++bin) {
    float re = 0.f, im = 0.f;
    for (int i2 = 0; i2 < 8; ++i2) {
      float ang = 6.28318530717958647f * (float)(bin * i2) * 0.125f;
      float v = noise[f * 8 + i2];
      re += v * __cosf(ang);
      im -= v * __sinf(ang);
    }
    nspec[f * 10 + bin * 2]     = re;
    nspec[f * 10 + bin * 2 + 1] = im;
  }
}

__global__ void k_audio(const float* __restrict__ filt, const float* __restrict__ nspec,
                        float* __restrict__ imp) {
  int i = blockIdx.x * 256 + threadIdx.x;
  if (i >= 256 * 512) return;
  int n = i >> 9, f = i & 511;
  int fm = f & 31;
  int g = (fm == 0) ? 1 : ((fm == 31) ? 2 : 0);
  int k = f >> 5;
  const float* fr = filt + ((size_t)g * 4096 + n * 16 + k) * 5;
  float Xre[5], Xim[5];
  for (int bin = 0; bin < 5; ++bin) {
    Xre[bin] = nspec[f * 10 + bin * 2]     * fr[bin];
    Xim[bin] = nspec[f * 10 + bin * 2 + 1] * fr[bin];
  }
  for (int s = 0; s < 8; ++s) {
    float a = Xre[0];
    for (int bin = 1; bin < 4; ++bin) {
      float ang = 6.28318530717958647f * (float)(bin * s) * 0.125f;
      a += 2.f * (Xre[bin] * __cosf(ang) - Xim[bin] * __sinf(ang));
    }
    a += Xre[4] * ((s & 1) ? -1.f : 1.f);
    imp[(size_t)n * 4096 + f * 8 + s] = a * 0.125f;
  }
}

__global__ void k_amp(const float* __restrict__ emb, const float* __restrict__ w,
                      const float* __restrict__ b, float* __restrict__ amps) {
  int e = blockIdx.x * 256 + threadIdx.x;
  if (e >= 256) return;
  float a = b[0];
  for (int j = 0; j < 256; ++j) a += emb[(size_t)e * 256 + j] * w[j];
  amps[e] = fabsf(a);
}

__global__ __launch_bounds__(256) void k_softmax_bf(const float* __restrict__ X,
                                                    unsigned short* __restrict__ Y, int N) {
  __shared__ float sh[256];
  int r = blockIdx.x, tid = threadIdx.x;
  const float* x = X + (size_t)r * N;
  float mx = -3e38f;
  for (int c = tid; c < N; c += 256) mx = fmaxf(mx, x[c]);
  sh[tid] = mx; __syncthreads();
  for (int s = 128; s > 0; s >>= 1) { if (tid < s) sh[tid] = fmaxf(sh[tid], sh[tid + s]); __syncthreads(); }
  mx = sh[0]; __syncthreads();
  float sum = 0.f;
  for (int c = tid; c < N; c += 256) sum += __expf(x[c] - mx);
  sh[tid] = sum; __syncthreads();
  for (int s = 128; s > 0; s >>= 1) { if (tid < s) sh[tid] += sh[tid + s]; __syncthreads(); }
  float inv = 1.0f / sh[0];
  for (int c = tid; c < N; c += 256) Y[(size_t)r * N + c] = f2bfu(__expf(x[c] - mx) * inv);
}

// (imp ⊛ res + imp) * amp, then scatter-add (shift by t_e*256, scale by packed val)
// directly into dry[b] -> fuses the sparse "up" convolution and the event sum.
__global__ __launch_bounds__(256) void k_event_conv(const float* __restrict__ imp,
                                                    const float* __restrict__ res,
                                                    const float* __restrict__ amps,
                                                    const int* __restrict__ ev_idx,
                                                    const float* __restrict__ ev_val,
                                                    float* __restrict__ dry) {
  __shared__ float s_imp[4096];
  __shared__ float s_res[4608];
  int e  = blockIdx.y;
  int s0 = blockIdx.x * 512;
  int tid = threadIdx.x;
  const float* impe = imp + (size_t)e * 4096;
  #pragma unroll
  for (int l = 0; l < 4; ++l)
    async_b128(&s_imp[tid * 4 + l * 1024], &impe[tid * 4 + l * 1024]);
  const float* rese = res + (size_t)e * 32768;
  #pragma unroll
  for (int l = 0; l < 18; ++l) {
    int p = tid + l * 256;
    int q = s0 - 4095 + p;
    s_res[p] = (q >= 0 && q < 32768) ? rese[q] : 0.f;
  }
  async_wait0();
  __syncthreads();
  int i0 = tid * 2;
  float a0 = 0.f, a1 = 0.f;
  #pragma unroll 4
  for (int j = 0; j < 4096; ++j) {
    float w = s_imp[j];
    int base = 4095 + i0 - j;
    a0 += w * s_res[base];
    a1 += w * s_res[base + 1];
  }
  int s = s0 + i0;
  float amp = amps[e];
  float v0 = (a0 + ((s     < 4096) ? s_imp[s]     : 0.f)) * amp;
  float v1 = (a1 + ((s + 1 < 4096) ? s_imp[s + 1] : 0.f)) * amp;
  int b = e >> 5;
  int shift = (ev_idx[e] & 127) << 8;
  float val = ev_val[e];
  int p0 = s + shift, p1 = s + 1 + shift;
  if (p0 < 32768) atomicAdd(&dry[(size_t)b * 32768 + p0], v0 * val);
  if (p1 < 32768) atomicAdd(&dry[(size_t)b * 32768 + p1], v1 * val);
}

__global__ void k_reverb_mlp(const float* __restrict__ vctx,
                             const float* __restrict__ w0, const float* __restrict__ b0,
                             const float* __restrict__ lg, const float* __restrict__ lb,
                             const float* __restrict__ wr, const float* __restrict__ br,
                             const float* __restrict__ wm, const float* __restrict__ bm,
                             float* __restrict__ rooms, float* __restrict__ mix) {
  __shared__ float h[32]; __shared__ float tmp[32];
  int b = blockIdx.x, j = threadIdx.x;
  float a = b0[j];
  for (int k = 0; k < 32; ++k) a += w0[j * 32 + k] * vctx[b * 32 + k];
  h[j] = a; __syncthreads();
  if (j == 0) {
    float m = 0.f; for (int k = 0; k < 32; ++k) m += h[k]; m /= 32.f;
    float v = 0.f; for (int k = 0; k < 32; ++k) { float d = h[k] - m; v += d * d; } v /= 32.f;
    tmp[0] = m; tmp[1] = rsqrtf(v + 1e-5f);
  }
  __syncthreads();
  float y = (h[j] - tmp[0]) * tmp[1] * lg[j] + lb[j];
  if (y < 0.f) y *= 0.2f;
  __syncthreads();
  h[j] = y; __syncthreads();
  if (j < 8) { float r = br[j]; for (int k = 0; k < 32; ++k) r += wr[j * 32 + k] * h[k]; tmp[8 + j] = r; }
  if (j < 2) { float r = bm[j]; for (int k = 0; k < 32; ++k) r += wm[j * 32 + k] * h[k]; tmp[16 + j] = r; }
  __syncthreads();
  if (j == 0) {
    float mx = -3e38f; for (int k = 0; k < 8; ++k) mx = fmaxf(mx, tmp[8 + k]);
    float s = 0.f; for (int k = 0; k < 8; ++k) { float e = __expf(tmp[8 + k] - mx); tmp[8 + k] = e; s += e; }
    for (int k = 0; k < 8; ++k) rooms[b * 8 + k] = tmp[8 + k] / s;
    float m0 = fmaxf(tmp[16], tmp[17]);
    float e0 = __expf(tmp[16] - m0), e1 = __expf(tmp[17] - m0);
    mix[b * 2] = e0 / (e0 + e1); mix[b * 2 + 1] = e1 / (e0 + e1);
  }
}

__global__ void k_rimp(const float* __restrict__ rooms, const float* __restrict__ irs,
                       float* __restrict__ rimp) {
  int i = blockIdx.x * 256 + threadIdx.x;
  if (i >= 8 * 32768) return;
  int b = i >> 15, s = i & 32767;
  float a = 0.f;
  for (int r = 0; r < 8; ++r) a += rooms[b * 8 + r] * irs[(size_t)r * 32768 + s];
  rimp[i] = a;
}

__global__ __launch_bounds__(256) void k_wet_final(const float* __restrict__ dry,
                                                   const float* __restrict__ rimp,
                                                   const float* __restrict__ mix,
                                                   float* __restrict__ outp) {
  __shared__ float sd[512]; __shared__ float si_[1024];
  int b  = blockIdx.y;
  int s0 = blockIdx.x * 512;
  int tid = threadIdx.x;
  int i0 = tid * 2;
  float a0 = 0.f, a1 = 0.f;
  const float* db = dry  + (size_t)b * 32768;
  const float* rb = rimp + (size_t)b * 32768;
  for (int jt = 0; jt < 64; ++jt) {
    __syncthreads();
    int J0 = jt * 512;
    sd[tid]       = db[J0 + tid];
    sd[tid + 256] = db[J0 + tid + 256];
    #pragma unroll
    for (int l = 0; l < 4; ++l) {
      int p = tid + l * 256;
      int q = s0 - J0 - 511 + p;
      si_[p] = (q >= 0 && q < 32768) ? rb[q] : 0.f;
    }
    __syncthreads();
    #pragma unroll 4
    for (int j = 0; j < 512; ++j) {
      float w = sd[j];
      int base = i0 + 511 - j;
      a0 += w * si_[base];
      a1 += w * si_[base + 1];
    }
  }
  int s = s0 + i0;
  float m0 = mix[b * 2], m1 = mix[b * 2 + 1];
  outp[(size_t)b * 32768 + s]     = db[s]     * m0 + a0 * m1;
  outp[(size_t)b * 32768 + s + 1] = db[s + 1] * m0 + a1 * m1;
}

// ---------------------------------------------------------------------------

extern "C" void kernel_launch(void* const* d_in, const int* in_sizes, int n_in,
                              void* d_out, int out_size, void* d_ws, size_t ws_size,
                              hipStream_t stream) {
  (void)in_sizes; (void)n_in; (void)out_size; (void)ws_size;
  int pi = 0;
  const float* X = (const float*)d_in[pi++];
  const float *encw[4], *encb[4], *encg[4], *encbeta[4];
  for (int i = 0; i < 4; ++i) {
    encw[i]    = (const float*)d_in[pi++]; encb[i]    = (const float*)d_in[pi++];
    encg[i]    = (const float*)d_in[pi++]; encbeta[i] = (const float*)d_in[pi++];
  }
  const float* enc_out_w = (const float*)d_in[pi++];
  const float* enc_out_b = (const float*)d_in[pi++];
  const float* ec_w = (const float*)d_in[pi++]; const float* ec_b = (const float*)d_in[pi++];
  const float* eo_w = (const float*)d_in[pi++]; const float* eo_b = (const float*)d_in[pi++];
  const float* vc_w = (const float*)d_in[pi++]; const float* vc_b = (const float*)d_in[pi++];
  const float* amp_w = (const float*)d_in[pi++]; const float* amp_b = (const float*)d_in[pi++];
  const float* tf_lin_w = (const float*)d_in[pi++]; const float* tf_lin_b = (const float*)d_in[pi++];
  const float *tfw[2], *tfb[2], *tfg[2], *tfbeta[2];
  for (int i = 0; i < 2; ++i) {
    tfw[i] = (const float*)d_in[pi++]; tfb[i]    = (const float*)d_in[pi++];
    tfg[i] = (const float*)d_in[pi++]; tfbeta[i] = (const float*)d_in[pi++];
  }
  const float* tf_out_w = (const float*)d_in[pi++]; const float* tf_out_b = (const float*)d_in[pi++];
  const float* nm_conv_w = (const float*)d_in[pi++]; const float* nm_conv_b = (const float*)d_in[pi++];
  const float* nm_spec_w = (const float*)d_in[pi++]; const float* nm_spec_b = (const float*)d_in[pi++];
  const float* res_sel_w = (const float*)d_in[pi++]; const float* res_sel_b = (const float*)d_in[pi++];
  const float* atoms = (const float*)d_in[pi++];
  pi += 10; // mx_* params: softmax over singleton axis => identity mix, unused
  const float* vb_l0_w = (const float*)d_in[pi++]; const float* vb_l0_b = (const float*)d_in[pi++];
  const float* vb_ln_g = (const float*)d_in[pi++]; const float* vb_ln_b = (const float*)d_in[pi++];
  const float* vb_rooms_w = (const float*)d_in[pi++]; const float* vb_rooms_b = (const float*)d_in[pi++];
  const float* vb_mix_w = (const float*)d_in[pi++]; const float* vb_mix_b = (const float*)d_in[pi++];
  const float* vb_room_irs = (const float*)d_in[pi++];
  const float* noise = (const float*)d_in[pi++];

  // ---- workspace carve ----
  char* wp = (char*)d_ws;
  auto carve = [&](size_t bytes) { char* p = wp; wp += (bytes + 255) & ~(size_t)255; return p; };
  unsigned short* Abuf   = (unsigned short*)carve((size_t)12288 * 1024);        // 12 MB bf16 A/im2col
  float* spec2  = (float*)carve((size_t)1024 * 2048 * 4);
  float* actA   = (float*)carve((size_t)1024 * 1024 * 4);
  float* actB   = (float*)carve((size_t)1024 * 1024 * 4);
  float* dense  = (float*)carve((size_t)1024 * 4096 * 4);
  float* denseF = (float*)carve((size_t)1024 * 4096 * 4);
  float* bn     = (float*)carve((size_t)2 * 4096 * 4);
  float* ctxt0  = (float*)carve((size_t)8 * 4096 * 4);
  float* ce     = (float*)carve((size_t)8 * 256 * 4);
  float* vctx   = (float*)carve((size_t)8 * 32 * 4);
  int*   ev_idx = (int*)carve((size_t)256 * 4);
  float* ev_val = (float*)carve((size_t)256 * 4);
  float* emb    = (float*)carve((size_t)256 * 256 * 4);
  unsigned short* emb_bf = (unsigned short*)carve((size_t)256 * 256 * 2);
  float* tfA    = (float*)carve((size_t)4096 * 128 * 4);
  float* tfB    = (float*)carve((size_t)4096 * 128 * 4);
  float* nmh    = (float*)carve((size_t)12288 * 128 * 4);
  float* filt   = (float*)carve((size_t)12288 * 5 * 4);
  float* nspec  = (float*)carve((size_t)512 * 10 * 4);
  float* impb   = (float*)carve((size_t)256 * 4096 * 4);
  float* amps   = (float*)carve((size_t)256 * 4);
  float* logits = (float*)carve((size_t)256 * 1024 * 4);
  unsigned short* selbf = (unsigned short*)carve((size_t)256 * 1024 * 2);
  float* resb   = (float*)carve((size_t)256 * 32768 * 4);                       // 32 MB
  float* dry    = (float*)carve((size_t)8 * 32768 * 4);
  float* rimp   = (float*)carve((size_t)8 * 32768 * 4);
  float* rooms  = (float*)carve((size_t)8 * 8 * 4);
  float* mixv   = (float*)carve((size_t)8 * 2 * 4);

  float* out_final = (float*)d_out;
  float* out_enc   = (float*)d_out + 262144;

  const float PI2 = 6.28318530717958647f;

  // ---- STFT magnitude as DFT GEMM ----
  k_build_frames<<<(1024 * 2048) / 256, 256, 0, stream>>>(X, Abuf);
  gemm_bf16<3><<<dim3(16, 8), 256, 0, stream>>>(Abuf, nullptr, spec2, nullptr,
                                                1024, 2048, 2048, 2048, 0, 0, PI2 / 2048.0f);
  k_mag<<<(1024 * 1024) / 256, 256, 0, stream>>>(spec2, actA);

  // ---- encoder: 4 dilated conv layers (WMMA GEMM) + BN ----
  const int dil[4] = {1, 3, 9, 1};
  float* cur = actA; float* nxt = actB;
  for (int i = 0; i < 4; ++i) {
    k_im2col_enc<<<(1024 * 3072) / 256, 256, 0, stream>>>(cur, Abuf, dil[i]);
    gemm_bf16<1><<<dim3(8, 8), 256, 0, stream>>>(Abuf, encw[i], nxt, encb[i],
                                                 1024, 1024, 3072, 1024, 3, 2, 0.f);
    k_bn_stats<<<1024, 256, 0, stream>>>(nxt, bn, 1024, 1024);
    k_bn_norm<<<(1024 * 1024) / 256, 256, 0, stream>>>(nxt, bn, encg[i], encbeta[i], 1024, 1024, 0);
    float* t = cur; cur = nxt; nxt = t;
  }
  // enc_out (1x1 conv)
  k_f2bf<<<(1024 * 1024) / 256, 256, 0, stream>>>(cur, Abuf, 1024 * 1024);
  gemm_bf16<2><<<dim3(32, 8), 256, 0, stream>>>(Abuf, enc_out_w, dense, enc_out_b,
                                                1024, 4096, 1024, 1024, 0, 1, 0.f);
  // refractory FIR
  k_fir<<<(1024 * 4096) / 256, 256, 0, stream>>>(dense, denseF);

  // ---- context + top-k ----
  k_ctxt_mean<<<(8 * 4096) / 256, 256, 0, stream>>>(denseF, ctxt0);
  k_gemv<<<8, 256, 0, stream>>>(ec_w, ctxt0, ec_b, ce, 8, 4096, 256);
  k_gemv<<<1, 256, 0, stream>>>(vc_w, ctxt0, vc_b, vctx, 8, 4096, 32);
  k_zero<<<(4194304 + 255) / 256, 256, 0, stream>>>(out_enc, 4194304);
  k_zero<<<(262144 + 255) / 256, 256, 0, stream>>>(dry, 262144);
  k_topk<<<8, 1024, 0, stream>>>(denseF, ev_idx, ev_val);
  k_scatter_enc<<<1, 256, 0, stream>>>(ev_idx, ev_val, out_enc);
  k_emb<<<(65536) / 256, 256, 0, stream>>>(ce, eo_w, eo_b, ev_idx, emb);
  k_f2bf<<<(65536) / 256, 256, 0, stream>>>(emb, emb_bf, 65536);

  // ---- impulse generator ----
  gemm_bf16<2><<<dim3(4, 2), 256, 0, stream>>>(emb_bf, tf_lin_w, tfA, tf_lin_b,
                                               256, 512, 256, 256, 0, 1, 0.f);
  // layer 0: T 4 -> 8
  k_tf_im2col<<<(2048 * 384) / 256, 256, 0, stream>>>(tfA, Abuf, 4, 8, 1, 0);
  gemm_bf16<1><<<dim3(1, 16), 256, 0, stream>>>(Abuf, tfw[0], tfB, tfb[0],
                                                2048, 128, 384, 128, 3, 1, 0.f);
  k_bn_stats<<<128, 256, 0, stream>>>(tfB, bn, 2048, 128);
  k_bn_norm<<<(2048 * 128) / 256, 256, 0, stream>>>(tfB, bn, tfg[0], tfbeta[0], 2048, 128, 1);
  // layer 1: T 8 -> 16
  k_tf_im2col<<<(4096 * 384) / 256, 256, 0, stream>>>(tfB, Abuf, 8, 16, 1, 1);
  gemm_bf16<1><<<dim3(1, 32), 256, 0, stream>>>(Abuf, tfw[1], tfA, tfb[1],
                                                4096, 128, 384, 128, 3, 1, 0.f);
  k_bn_stats<<<128, 256, 0, stream>>>(tfA, bn, 4096, 128);
  k_bn_norm<<<(4096 * 128) / 256, 256, 0, stream>>>(tfA, bn, tfg[1], tfbeta[1], 4096, 128, 1);
  // tf_out: T 16 -> 16
  k_tf_im2col<<<(4096 * 384) / 256, 256, 0, stream>>>(tfA, Abuf, 16, 16, 0, 1);
  gemm_bf16<1><<<dim3(1, 32), 256, 0, stream>>>(Abuf, tf_out_w, tfB, tf_out_b,
                                                4096, 128, 384, 128, 3, 1, 0.f);
  // nm_conv on collapsed 3-group representation (exact: input is 32x repeat)
  k_nm_im2col<<<(12288 * 384) / 256, 256, 0, stream>>>(tfB, Abuf);
  gemm_bf16<1><<<dim3(1, 96), 256, 0, stream>>>(Abuf, nm_conv_w, nmh, nm_conv_b,
                                                12288, 128, 384, 128, 3, 2, 0.f);
  k_nmspec<<<48, 256, 0, stream>>>(nmh, nm_spec_w, nm_spec_b, filt);
  k_nspec<<<2, 256, 0, stream>>>(noise, nspec);
  k_audio<<<(256 * 512) / 256, 256, 0, stream>>>(filt, nspec, impb);
  k_amp<<<1, 256, 0, stream>>>(emb, amp_w, amp_b, amps);

  // ---- resonance: softmax(emb @ res_sel) @ atoms (WMMA) ----
  gemm_bf16<2><<<dim3(8, 2), 256, 0, stream>>>(emb_bf, res_sel_w, logits, res_sel_b,
                                               256, 1024, 256, 256, 0, 1, 0.f);
  k_softmax_bf<<<256, 256, 0, stream>>>(logits, selbf, 1024);
  gemm_bf16<0><<<dim3(256, 2), 256, 0, stream>>>(selbf, atoms, resb, nullptr,
                                                 256, 32768, 1024, 32768, 0, 0, 0.f);

  // ---- per-event conv + amp + shifted scatter into dry (mix MLP == identity) ----
  k_event_conv<<<dim3(64, 256), 256, 0, stream>>>(impb, resb, amps, ev_idx, ev_val, dry);

  // ---- reverb ----
  k_reverb_mlp<<<8, 32, 0, stream>>>(vctx, vb_l0_w, vb_l0_b, vb_ln_g, vb_ln_b,
                                     vb_rooms_w, vb_rooms_b, vb_mix_w, vb_mix_b, rooms, mixv);
  k_rimp<<<(8 * 32768) / 256, 256, 0, stream>>>(rooms, vb_room_irs, rimp);
  k_wet_final<<<dim3(64, 8), 256, 0, stream>>>(dry, rimp, mixv, out_final);
}
